// HeteroAttentionLayer_53695681134796
// MI455X (gfx1250) — matrix operand order repaired and checked by reading, not compiled, not run on hardware
//
#include <hip/hip_runtime.h>

// ---------------------------------------------------------------------------
// Types
// ---------------------------------------------------------------------------
typedef __bf16 bhalf8_t  __attribute__((ext_vector_type(8)));
typedef __bf16 bhalf16_t __attribute__((ext_vector_type(16)));
typedef float  floatx8   __attribute__((ext_vector_type(8)));
typedef unsigned int u32x4 __attribute__((ext_vector_type(4)));
typedef int          i32x8 __attribute__((ext_vector_type(8)));
typedef int          i32x4 __attribute__((ext_vector_type(4)));

static constexpr int HID   = 1024;   // HIDDEN
static constexpr int NHEAD = 16;
static constexpr int HDIM  = 64;

// GEMM blocking
static constexpr int BM = 64;    // rows per block (4 WMMA row tiles)
static constexpr int BN = 128;   // cols per block (8 WMMA col tiles)
static constexpr int KC = 64;    // K chunk staged in LDS
static constexpr int LDS_COL_STRIDE = 72;  // 64 halves + 8 pad halves (bank spread)

#if __has_builtin(__builtin_amdgcn_tensor_load_to_lds) && \
    __has_builtin(__builtin_amdgcn_s_wait_tensorcnt)
#define USE_TDM 1
#else
#define USE_TDM 0
#endif

__device__ __forceinline__ unsigned short f32_to_bf16_rne(float f) {
    unsigned u = __float_as_uint(f);
    unsigned r = (u + 0x7FFFu + ((u >> 16) & 1u)) >> 16;
    return (unsigned short)r;
}

__device__ __forceinline__ bhalf16_t cat16(bhalf8_t lo, bhalf8_t hi) {
    return __builtin_shufflevector(lo, hi, 0, 1, 2, 3, 4, 5, 6, 7,
                                           8, 9, 10, 11, 12, 13, 14, 15);
}

// ---------------------------------------------------------------------------
// f32 -> bf16 elementwise conversion
// ---------------------------------------------------------------------------
__global__ void cvt_bf16_kernel(const float* __restrict__ in,
                                unsigned short* __restrict__ out, int n) {
    int i = blockIdx.x * blockDim.x + threadIdx.x;
    if (i < n) out[i] = f32_to_bf16_rne(in[i]);
}

// ---------------------------------------------------------------------------
// Weight transpose + convert: W[t][k][n] (f32) -> Wt[t][n][k] (bf16)
// block (32,8), grid (N/32, K/32, T)
// ---------------------------------------------------------------------------
__global__ void transpose_cvt_kernel(const float* __restrict__ W,
                                     unsigned short* __restrict__ Wt) {
    __shared__ float tile[32][33];
    const int t  = blockIdx.z;
    const int n0 = blockIdx.x * 32;
    const int k0 = blockIdx.y * 32;
    const int lx = threadIdx.x;   // 0..31
    const int ly = threadIdx.y;   // 0..7
    const size_t mat = (size_t)t * HID * HID;
#pragma unroll
    for (int i = 0; i < 32; i += 8)
        tile[ly + i][lx] = W[mat + (size_t)(k0 + ly + i) * HID + (n0 + lx)];
    __syncthreads();
#pragma unroll
    for (int i = 0; i < 32; i += 8)
        Wt[mat + (size_t)(n0 + ly + i) * HID + (k0 + lx)] =
            f32_to_bf16_rne(tile[lx][ly + i]);
}

// ---------------------------------------------------------------------------
// Zero the softmax max/denominator buffers
// ---------------------------------------------------------------------------
__global__ void init_softmax_kernel(unsigned* __restrict__ maxenc,
                                    float* __restrict__ denom, int n) {
    int i = blockIdx.x * blockDim.x + threadIdx.x;
    if (i < n) { maxenc[i] = 0u; denom[i] = 0.0f; }
}

// ---------------------------------------------------------------------------
// Typed-block GEMM:  out[M x 1024] = A[M x 1024] @ W[type(row)] + b[type(row)]
//   A   : bf16 row-major.  Wt : bf16 column-major (Wt[t][n][k]), T stacked.
// Block = 256 threads (8 waves) computes a 64x128 output tile.
// Wave w: mi = w>>1 (row tile), ni = w&1 (64-col half); 4 accumulators each.
// B tiles (128 cols x 64 K) are double-buffered in LDS and filled by the
// Tensor Data Mover with descriptor padding (interval 32 dwords, pad 4 dwords)
// -> 72-half column stride so B-fragment ds reads hit 16 distinct banks.
// TENSORcnt completes in order per wave, so waiting tensorcnt<=1 after
// issuing chunk i+1 guarantees chunk i is resident -> DMA overlaps compute.
// grid = (1024/BN, M/BM)
// ---------------------------------------------------------------------------
__global__ void gemm_bf16_wmma_kernel(const unsigned short* __restrict__ A,
                                      const unsigned short* __restrict__ Wt,
                                      const float* __restrict__ bias,
                                      float* __restrict__ out,
                                      int typeBound1, int typeBound2) {
    __shared__ __align__(16) unsigned short Bsh[2][BN * LDS_COL_STRIDE]; // 36 KB

    const int lane  = threadIdx.x & 31;
    const int wv    = threadIdx.x >> 5;
    const int mi    = wv >> 1;           // 0..3
    const int ni    = wv & 1;            // 0..1
    const int half  = lane >> 4;         // 0: lanes 0-15, 1: lanes 16-31
    const int tileM = blockIdx.y * BM + mi * 16;
    const int n0    = blockIdx.x * BN;   // first column of the block
    const int t     = (tileM >= typeBound1) + (tileM >= typeBound2);

    // A fragment per ISA table: lanes 0-15 comps = K 0..7 | 16..23,
    // lanes 16-31 comps = K 8..15 | 24..31.
    const unsigned short* Ar =
        A + (size_t)(tileM + (lane & 15)) * HID + half * 8;
    const unsigned short* Wtm = Wt + (size_t)t * HID * HID;

    floatx8 acc[4] = {};
    const int NCH = HID / KC;            // 16 K-chunks

#if USE_TDM
    auto tdm_issue = [&](int kk, int buf) {
        unsigned long long ga =
            (unsigned long long)(const void*)(Wtm + (size_t)n0 * HID + kk);
        unsigned lds_base = (unsigned)(size_t)(const void*)&Bsh[buf][0];
        u32x4 g0 = { 1u,                                   // count=1 (valid)
                     lds_base,                             // lds_addr
                     (unsigned)ga,                         // global_addr lo
                     (unsigned)((ga >> 32) & 0x1FFFFFFu) | (2u << 30) }; // hi|type=2
        // dw0: data_size=2B(code1)<<16 | pad_enable<<20 |
        //      pad_interval(code4:32dw)<<22 | pad_amount(code3:4dw)<<25
        i32x8 g1 = { (int)0x07110000u,
                     (int)(1024u << 16),        // tensor_dim0 = 1024 (K)
                     (int)(128u  << 16),        // tensor_dim1 = 128 (cols)
                     (int)((unsigned)KC << 16), // tile_dim0 = 64
                     (int)BN,                   // tile_dim1 = 128
                     (int)HID,                  // tensor_dim0_stride = 1024
                     0, 0 };
        i32x4 gz4 = { 0, 0, 0, 0 };
        i32x8 gz8 = { 0, 0, 0, 0, 0, 0, 0, 0 };
        __builtin_amdgcn_tensor_load_to_lds(g0, g1, gz4, gz4, gz8, 0);
    };
    if (wv == 0) tdm_issue(0, 0);
#endif

    for (int ci = 0; ci < NCH; ++ci) {
        const int kk = ci * KC;
#if USE_TDM
        if (wv == 0) {
            if (ci + 1 < NCH) {
                tdm_issue(kk + KC, (ci + 1) & 1);     // prefetch next chunk
                __builtin_amdgcn_s_wait_tensorcnt(1); // chunk ci resident
            } else {
                __builtin_amdgcn_s_wait_tensorcnt(0);
            }
        }
        __syncthreads();   // buffer ci&1 visible to all waves
        const unsigned short* bsh = &Bsh[ci & 1][0];
#else
        __syncthreads();   // previous chunk fully consumed
        // Fallback: cooperative copy with identical padded layout.
        for (int idx = threadIdx.x; idx < BN * (KC / 16); idx += 256) {
            const int c = idx >> 2, s = idx & 3;
            const unsigned short* src = Wtm + (size_t)(n0 + c) * HID + kk + s * 16;
            unsigned short* dst = &Bsh[0][c * LDS_COL_STRIDE + s * 16];
            *(bhalf8_t*)dst       = *(const bhalf8_t*)src;
            *(bhalf8_t*)(dst + 8) = *(const bhalf8_t*)(src + 8);
        }
        __syncthreads();
        const unsigned short* bsh = &Bsh[0][0];
#endif

#pragma unroll
        for (int k2 = 0; k2 < KC; k2 += 32) {
            // Load all 4 B fragments first (one ds clause, one wait), then
            // the A fragment, then 4 back-to-back WMMAs.
            bhalf16_t bfr[4];
#pragma unroll
            for (int j = 0; j < 4; ++j) {
                const int cl = ni * 64 + j * 16 + (lane & 15);
                const unsigned short* Bp =
                    &bsh[cl * LDS_COL_STRIDE + k2 + half * 8];
                bfr[j] = cat16(*(const bhalf8_t*)Bp,
                               *(const bhalf8_t*)(Bp + 16));
            }
            bhalf8_t alo = *(const bhalf8_t*)(Ar + kk + k2);
            bhalf8_t ahi = *(const bhalf8_t*)(Ar + kk + k2 + 16);
            bhalf16_t a = cat16(alo, ahi);
#pragma unroll
            for (int j = 0; j < 4; ++j)
                acc[j] = __builtin_amdgcn_wmma_f32_16x16x32_bf16(
                    false, a, false, bfr[j], (short)0, acc[j], false, false);
        }
        __syncthreads();   // chunk consumed; safe to overwrite this buffer
    }

    // D layout: VGPR r -> row tileM + r + 8*half, lane&15 -> column
    const int rbase = tileM + half * 8;
#pragma unroll
    for (int j = 0; j < 4; ++j) {
        const int col = n0 + ni * 64 + j * 16 + (lane & 15);
        const float bv = bias[t * HID + col];
#pragma unroll
        for (int r = 0; r < 8; ++r)
            out[(size_t)(rbase + r) * HID + col] = acc[j][r] + bv;
    }
}

// ---------------------------------------------------------------------------
// Pass A: per-edge per-head attention scores + segment max (order-preserving
// uint encoding + atomicMax). One wave per edge; lanes parallel over head_dim.
// ---------------------------------------------------------------------------
__global__ void edge_scores_kernel(const float* __restrict__ w3,
                                   const float* __restrict__ w4,
                                   const float* __restrict__ w5,
                                   const int* __restrict__ esrc,
                                   const int* __restrict__ edst,
                                   float* __restrict__ scores,
                                   unsigned* __restrict__ maxenc, int E) {
    const int e = blockIdx.x * (blockDim.x >> 5) + (threadIdx.x >> 5);
    const int lane = threadIdx.x & 31;
    if (e >= E) return;
    const int s = esrc[e], d = edst[e];
    const float* qp = w3 + (size_t)d * HID;
    const float* kp = w4 + (size_t)s * HID;
    const float* ep = w5 + (size_t)e * HID;
    for (int h = 0; h < NHEAD; ++h) {
        const int o = h * HDIM + lane;
        float v = qp[o] * (kp[o] + ep[o]) +
                  qp[o + 32] * (kp[o + 32] + ep[o + 32]);
#pragma unroll
        for (int off = 16; off > 0; off >>= 1) v += __shfl_xor(v, off, 32);
        if (lane == 0) {
            const float sc = v * 0.125f;   // 1/sqrt(64)
            scores[(size_t)e * NHEAD + h] = sc;
            unsigned m = __float_as_uint(sc);
            m = (m & 0x80000000u) ? ~m : (m | 0x80000000u);
            atomicMax(&maxenc[d * NHEAD + h], m);
        }
    }
}

// ---------------------------------------------------------------------------
// Pass B: ex = exp(score - max[dst]); denom[dst] += ex  (one thread per (e,h))
// ---------------------------------------------------------------------------
__global__ void edge_exp_kernel(const int* __restrict__ edst,
                                float* __restrict__ scores,
                                const unsigned* __restrict__ maxenc,
                                float* __restrict__ denom, int EH) {
    const int i = blockIdx.x * blockDim.x + threadIdx.x;
    if (i >= EH) return;
    const int e = i >> 4, h = i & 15;
    const int d = edst[e];
    const unsigned m = maxenc[d * NHEAD + h];
    const float mx =
        __uint_as_float((m & 0x80000000u) ? (m & 0x7FFFFFFFu) : ~m);
    const float ex = __expf(scores[i] - mx);
    scores[i] = ex;
    atomicAdd(&denom[d * NHEAD + h], ex);
}

// ---------------------------------------------------------------------------
// Pass C: out[dst] += (ex/denom) * (w2[src] + w5[e]). One wave per edge.
// d_out already holds w1, so this completes out = w1 + agg.
// ---------------------------------------------------------------------------
__global__ void edge_aggregate_kernel(const float* __restrict__ w2,
                                      const float* __restrict__ w5,
                                      const int* __restrict__ esrc,
                                      const int* __restrict__ edst,
                                      const float* __restrict__ scores,
                                      const float* __restrict__ denom,
                                      float* __restrict__ out, int E) {
    const int e = blockIdx.x * (blockDim.x >> 5) + (threadIdx.x >> 5);
    const int lane = threadIdx.x & 31;
    if (e >= E) return;
    const int s = esrc[e], d = edst[e];
    const float* vp = w2 + (size_t)s * HID;
    const float* ep = w5 + (size_t)e * HID;
    float* op = out + (size_t)d * HID;
    for (int h = 0; h < NHEAD; ++h) {
        const float attn =
            scores[(size_t)e * NHEAD + h] / denom[d * NHEAD + h];
        const int o = h * HDIM + lane;
        atomicAdd(&op[o], attn * (vp[o] + ep[o]));
        atomicAdd(&op[o + 32], attn * (vp[o + 32] + ep[o + 32]));
    }
}

// ---------------------------------------------------------------------------
// Host-side launch
// ---------------------------------------------------------------------------
extern "C" void kernel_launch(void* const* d_in, const int* in_sizes, int n_in,
                              void* d_out, int out_size, void* d_ws,
                              size_t ws_size, hipStream_t stream) {
    const float* X    = (const float*)d_in[0];
    const float* Ef   = (const float*)d_in[1];
    const int*   esrc = (const int*)d_in[2];
    const int*   edst = (const int*)d_in[3];
    const float* W1 = (const float*)d_in[4];  const float* b1 = (const float*)d_in[5];
    const float* W2 = (const float*)d_in[6];  const float* b2 = (const float*)d_in[7];
    const float* W3 = (const float*)d_in[8];  const float* b3 = (const float*)d_in[9];
    const float* W4 = (const float*)d_in[10]; const float* b4 = (const float*)d_in[11];
    const float* W5 = (const float*)d_in[12]; const float* b5 = (const float*)d_in[13];

    const int Nn = in_sizes[0] / HID;   // 8192 nodes
    const int E  = in_sizes[2];         // 65536 edges
    float* out = (float*)d_out;

    // --- carve workspace ---------------------------------------------------
    char* p = (char*)d_ws;
    auto carve = [&](size_t bytes) {
        char* r = p;
        p += (bytes + 255) & ~(size_t)255;
        return (void*)r;
    };
    unsigned short* Xb  = (unsigned short*)carve((size_t)Nn * HID * 2);
    unsigned short* Eb  = (unsigned short*)carve((size_t)E * HID * 2);
    unsigned short* Wt  = (unsigned short*)carve((size_t)14 * HID * HID * 2);
    float* w2buf = (float*)carve((size_t)Nn * HID * 4);
    float* w3buf = (float*)carve((size_t)Nn * HID * 4);
    float* w4buf = (float*)carve((size_t)Nn * HID * 4);
    float* w5buf = (float*)carve((size_t)E * HID * 4);
    float*    scores = (float*)carve((size_t)E * NHEAD * 4);
    unsigned* maxenc = (unsigned*)carve((size_t)Nn * NHEAD * 4);
    float*    denom  = (float*)carve((size_t)Nn * NHEAD * 4);

    const size_t MATE = (size_t)HID * HID;  // elements per weight matrix
    unsigned short* W1t = Wt + 0 * MATE;
    unsigned short* W2t = Wt + 3 * MATE;
    unsigned short* W3t = Wt + 6 * MATE;
    unsigned short* W4t = Wt + 9 * MATE;
    unsigned short* W5t = Wt + 12 * MATE;

    // --- 1) convert activations to bf16 ------------------------------------
    {
        int n = Nn * HID;
        cvt_bf16_kernel<<<(n + 255) / 256, 256, 0, stream>>>(X, Xb, n);
        n = E * HID;
        cvt_bf16_kernel<<<(n + 255) / 256, 256, 0, stream>>>(Ef, Eb, n);
    }

    // --- 2) transpose + convert weights ------------------------------------
    {
        dim3 blk(32, 8);
        transpose_cvt_kernel<<<dim3(HID / 32, HID / 32, 3), blk, 0, stream>>>(W1, W1t);
        transpose_cvt_kernel<<<dim3(HID / 32, HID / 32, 3), blk, 0, stream>>>(W2, W2t);
        transpose_cvt_kernel<<<dim3(HID / 32, HID / 32, 3), blk, 0, stream>>>(W3, W3t);
        transpose_cvt_kernel<<<dim3(HID / 32, HID / 32, 3), blk, 0, stream>>>(W4, W4t);
        transpose_cvt_kernel<<<dim3(HID / 32, HID / 32, 2), blk, 0, stream>>>(W5, W5t);
    }

    // --- 3) zero softmax state ---------------------------------------------
    {
        int n = Nn * NHEAD;
        init_softmax_kernel<<<(n + 255) / 256, 256, 0, stream>>>(maxenc, denom, n);
    }

    // --- 4) typed-block projection GEMMs (WMMA bf16 + TDM-staged B) ---------
    {
        const int NB1 = 4096, NB2 = 6144;        // node type boundaries
        const int EB1 = 32768, BIG = 0x7FFFFFFF; // edge type boundary
        dim3 gN(HID / BN, Nn / BM);
        dim3 gE(HID / BN, E / BM);
        gemm_bf16_wmma_kernel<<<gN, 256, 0, stream>>>(Xb, W1t, b1, out,   NB1, NB2);
        gemm_bf16_wmma_kernel<<<gN, 256, 0, stream>>>(Xb, W2t, b2, w2buf, NB1, NB2);
        gemm_bf16_wmma_kernel<<<gN, 256, 0, stream>>>(Xb, W3t, b3, w3buf, NB1, NB2);
        gemm_bf16_wmma_kernel<<<gN, 256, 0, stream>>>(Xb, W4t, b4, w4buf, NB1, NB2);
        gemm_bf16_wmma_kernel<<<gE, 256, 0, stream>>>(Eb, W5t, b5, w5buf, EB1, BIG);
    }

    // --- 5) attention scores + segment max ----------------------------------
    edge_scores_kernel<<<(E + 7) / 8, 256, 0, stream>>>(
        w3buf, w4buf, w5buf, esrc, edst, scores, maxenc, E);

    // --- 6) exp + segment sum -----------------------------------------------
    {
        int EH = E * NHEAD;
        edge_exp_kernel<<<(EH + 255) / 256, 256, 0, stream>>>(
            edst, scores, maxenc, denom, EH);
    }

    // --- 7) weighted aggregation onto d_out (which holds w1) ----------------
    edge_aggregate_kernel<<<(E + 7) / 8, 256, 0, stream>>>(
        w2buf, w5buf, esrc, edst, scores, denom, out, E);
}
